// GraphConvLayer_55748675502824
// MI455X (gfx1250) — matrix operand order repaired and checked by reading, not compiled, not run on hardware
//
#include <hip/hip_runtime.h>
#include <hip/hip_bf16.h>

// Problem constants (from reference): B=2, F_IN=32, T=8, N=20000, E=320000,
// HID=64, F_OUT=32.  BT = B*T = 16.
// Algebraic reorder: layer1 = (S x) W1 + b1  (== S (x W1) + b1, scatter on 32ch)
//                    layer2 = tanh(S (h1 W2) + b2)

typedef __attribute__((ext_vector_type(16))) _Float16 v16h;
typedef __attribute__((ext_vector_type(8)))  _Float16 v8h;
typedef __attribute__((ext_vector_type(8)))  float    v8f;

static constexpr int kB   = 2;
static constexpr int kT   = 8;
static constexpr int kFin = 32;
static constexpr int kNPW = 8;   // nodes per wave in the WMMA kernels
static constexpr int kEPW = 4;   // edges per wave in the scatter kernel

// ---------------------------------------------------------------- small ops
__global__ void fill_f32(float* p, float v, int n) {
  int i = blockIdx.x * 256 + threadIdx.x;
  if (i < n) p[i] = v;
}

__global__ void deg_kernel(const int* __restrict__ ei, const float* __restrict__ ew,
                           float* __restrict__ deg, int E) {
  int e = blockIdx.x * 256 + threadIdx.x;
  if (e < E) unsafeAtomicAdd(&deg[ei[E + e]], ew[e]);   // deg[col] += w
}

__global__ void dis_kernel(const float* __restrict__ deg, float* __restrict__ dis, int n) {
  int i = blockIdx.x * 256 + threadIdx.x;
  if (i < n) {
    float d = deg[i];
    dis[i] = (d > 0.f) ? rsqrtf(d) : 0.f;
  }
}

__global__ void norm_kernel(const int* __restrict__ ei, const float* __restrict__ ew,
                            const float* __restrict__ dis, float* __restrict__ nrm, int E) {
  int e = blockIdx.x * 256 + threadIdx.x;
  if (e < E) nrm[e] = dis[ei[e]] * ew[e] * dis[ei[E + e]];
}

// --------------------------------------------------- x[B,F,T,N] -> xT[n][bt][f] (f16)
// LDS-tiled: coalesced global reads over n, coalesced f16 writes over f.
__global__ void transpose_kernel(const float* __restrict__ x, _Float16* __restrict__ xT, int Nn) {
  __shared__ float lds[512 * 17];             // combo-major, padded
  const int n0 = blockIdx.x * 16;
  for (int j = 0; j < 32; ++j) {
    int idx = j * 256 + threadIdx.x;          // 0..8191
    int nl = idx & 15;
    int combo = idx >> 4;                     // combo = ((b*32+f)*8+t)
    if (n0 + nl < Nn) lds[combo * 17 + nl] = x[(size_t)combo * Nn + n0 + nl];
  }
  __syncthreads();
  for (int j = 0; j < 32; ++j) {
    int idx = j * 256 + threadIdx.x;
    int f  = idx & 31;
    int bt = (idx >> 5) & 15;
    int nl = idx >> 9;
    int b = bt >> 3, t = bt & 7;
    int combo = (b * 32 + f) * 8 + t;
    if (n0 + nl < Nn)
      xT[((size_t)(n0 + nl) * 16 + bt) * 32 + f] = (_Float16)lds[combo * 17 + nl];
  }
}

// ----------------------------------------------------------------- scatter
// One wave per kEPW edges: dst[col][bt][c] += norm * src[row][bt][c] (16x32 ch).
// Edges [E, E+N) are the self loops with weight dis[n]^2.  The next edge's
// source row is prefetched (global_prefetch_b8) while this edge's atomics issue.
__global__ void scatter_kernel(const _Float16* __restrict__ src, float* __restrict__ dst,
                               const int* __restrict__ ei, const float* __restrict__ nrm,
                               const float* __restrict__ dis, int E, int Nn) {
  int wave = (int)((blockIdx.x * (unsigned)blockDim.x + threadIdx.x) >> 5);
  int lane = threadIdx.x & 31;
  int total = E + Nn;
  int e0 = wave * kEPW;
  if (e0 >= total) return;
  int eend = (e0 + kEPW < total) ? e0 + kEPW : total;
  for (int e = e0; e < eend; ++e) {
    int r, c; float w;
    if (e < E) {
      r = ei[e]; c = ei[E + e]; w = nrm[e];
    } else {
      r = c = e - E;
      float d = dis[r];
      w = d * d;                               // self-loop weight 1.0 * dis^2
    }
    int en = e + 1;
    if (en < eend) {                           // prefetch next gather row (1 KB)
      int rn = (en < E) ? ei[en] : (en - E);
      __builtin_prefetch((const void*)(src + (size_t)rn * 512 + lane * 16), 0, 3);
    }
    const _Float16* sp = src + (size_t)r * 512;  // 16*32 contiguous per node
    float*          dp = dst + (size_t)c * 512;
#pragma unroll
    for (int i = 0; i < 16; ++i) {             // i == bt, lane == channel
      float v = (float)sp[i * 32 + lane] * w;
      unsafeAtomicAdd(dp + i * 32 + lane, v);  // global_atomic_add_f32
    }
  }
}

// ----------------------------------------------------------- WMMA fragments
// 16-bit A 16x32 (ISA 7.12.2): lanes 0-15: K 0..7 then 16..23; lanes 16-31: K 8..15 then 24..31.
__device__ inline v16h load_a_frag_f16(const _Float16* p, int lane) {
  int base = (lane >> 4) << 3;                 // 0 or 8
  v8h lo = *(const v8h*)(p + base);
  v8h hi = *(const v8h*)(p + base + 16);
  v16h a;
#pragma unroll
  for (int i = 0; i < 8; ++i) { a[i] = lo[i]; a[8 + i] = hi[i]; }
  return a;
}

__device__ inline v16h load_a_frag_f32(const float* p, int lane) {
  int base = (lane >> 4) << 3;
  v16h a;
#pragma unroll
  for (int i = 0; i < 8; ++i) {
    a[i]     = (_Float16)p[base + i];
    a[8 + i] = (_Float16)p[base + 16 + i];
  }
  return a;
}

// --------------------------------------------------------------- GEMM1 (32->64)
// W1 staged in LDS pre-swizzled into per-lane fragment order: each lane pulls its
// whole v16h B-fragment with contiguous 32B LDS loads.  Each wave keeps all 4
// B-fragments in registers and loops over kNPW nodes.
__global__ void gemm1_kernel(const float* __restrict__ aggX, const float* __restrict__ W1,
                             const float* __restrict__ b1, _Float16* __restrict__ h1, int Nn) {
  __shared__ _Float16 Wsw[4 * 32 * 16];        // [tile][lane][elem], elem i: k = 16*(i/8) + 8*(lane/16) + i%8
  for (int idx = threadIdx.x; idx < 2048; idx += 256) {
    int i = idx & 15;                          // fragment element
    int l = (idx >> 4) & 31;                   // lane
    int t = idx >> 9;                          // N-tile
    int k = ((i >> 3) << 4) + ((l >> 4) << 3) + (i & 7);  // lanes 16-31 hold K 8..15, 24..31
    int col = t * 16 + (l & 15);
    Wsw[idx] = (_Float16)W1[k * 64 + col];
  }
  __syncthreads();
  int lane = threadIdx.x & 31;
  int hi = lane >> 4, j = lane & 15;
  v16h bfrag[4];
#pragma unroll
  for (int t = 0; t < 4; ++t)
    bfrag[t] = *(const v16h*)&Wsw[(t * 32 + lane) * 16];
  int n0 = (blockIdx.x * 8 + (threadIdx.x >> 5)) * kNPW;
  for (int i = 0; i < kNPW; ++i) {
    int n = n0 + i;
    if (n >= Nn) break;                        // uniform per wave: EXEC stays full
    v16h a = load_a_frag_f32(aggX + ((size_t)n * 16 + j) * 32, lane);
#pragma unroll
    for (int t = 0; t < 4; ++t) {
      v8f c = {};
      c = __builtin_amdgcn_wmma_f32_16x16x32_f16(false, a, false, bfrag[t], (short)0, c,
                                                 false, false);
      int ch = t * 16 + j;
      float bias = b1[ch];
#pragma unroll
      for (int r = 0; r < 8; ++r)              // C/D: VGPR r -> row r (+8 for hi half)
        h1[((size_t)n * 16 + r + hi * 8) * 64 + ch] = (_Float16)(c[r] + bias);
    }
  }
}

// --------------------------------------------------------------- GEMM2 (64->32)
// K=64 as two chained 16x16x32 WMMAs; no bias (b2 added after scatter).
__global__ void gemm2_kernel(const _Float16* __restrict__ h1, const float* __restrict__ W2,
                             _Float16* __restrict__ g2, int Nn) {
  __shared__ _Float16 Wsw[4 * 32 * 16];        // [kchunk*2+tile][lane][elem]
  for (int idx = threadIdx.x; idx < 2048; idx += 256) {
    int i = idx & 15;
    int l = (idx >> 4) & 31;
    int f = idx >> 9;                          // 0..3
    int kc = f >> 1, t = f & 1;
    int k = kc * 32 + ((i >> 3) << 4) + ((l >> 4) << 3) + (i & 7);
    int col = t * 16 + (l & 15);
    Wsw[idx] = (_Float16)W2[k * 32 + col];
  }
  __syncthreads();
  int lane = threadIdx.x & 31;
  int hi = lane >> 4, j = lane & 15;
  v16h bfrag[4];
#pragma unroll
  for (int f = 0; f < 4; ++f)
    bfrag[f] = *(const v16h*)&Wsw[(f * 32 + lane) * 16];
  int n0 = (blockIdx.x * 8 + (threadIdx.x >> 5)) * kNPW;
  for (int i = 0; i < kNPW; ++i) {
    int n = n0 + i;
    if (n >= Nn) break;
    const _Float16* row = h1 + ((size_t)n * 16 + j) * 64;
    v16h a0 = load_a_frag_f16(row, lane);      // K 0..31
    v16h a1 = load_a_frag_f16(row + 32, lane); // K 32..63
#pragma unroll
    for (int t = 0; t < 2; ++t) {
      v8f c = {};
      c = __builtin_amdgcn_wmma_f32_16x16x32_f16(false, a0, false, bfrag[t], (short)0, c,
                                                 false, false);
      c = __builtin_amdgcn_wmma_f32_16x16x32_f16(false, a1, false, bfrag[2 + t], (short)0, c,
                                                 false, false);
      int ch = t * 16 + j;
#pragma unroll
      for (int r = 0; r < 8; ++r)
        g2[((size_t)n * 16 + r + hi * 8) * 32 + ch] = (_Float16)c[r];
    }
  }
}

// ------------------------------------------- out[B,C,T,N] = tanh(agg2 + b2), LDS tiled
__global__ void final_kernel(const float* __restrict__ agg2, const float* __restrict__ b2,
                             float* __restrict__ out, int Nn) {
  __shared__ float lds[16 * 513];
  const int n0 = blockIdx.x * 16;
  for (int j = 0; j < 32; ++j) {
    int idx = j * 256 + threadIdx.x;           // 0..8191
    int nl = idx >> 9;
    int q  = idx & 511;                        // bt*32 + c
    if (n0 + nl < Nn) lds[nl * 513 + q] = agg2[(size_t)(n0) * 512 + idx];
  }
  __syncthreads();
  for (int j = 0; j < 32; ++j) {
    int idx = j * 256 + threadIdx.x;
    int nl = idx & 15;
    int combo = idx >> 4;                      // ((b*32+c)*8+t)
    int b = combo >> 8;
    int c = (combo >> 3) & 31;
    int t = combo & 7;
    int bt = b * 8 + t;
    if (n0 + nl < Nn)
      out[(size_t)combo * Nn + n0 + nl] = tanhf(lds[nl * 513 + bt * 32 + c] + b2[c]);
  }
}

// ------------------------------------------------------------------ launch
extern "C" void kernel_launch(void* const* d_in, const int* in_sizes, int n_in,
                              void* d_out, int out_size, void* d_ws, size_t ws_size,
                              hipStream_t stream) {
  const float* x  = (const float*)d_in[0];
  const int*   ei = (const int*)d_in[1];
  const float* ew = (const float*)d_in[2];
  const float* W1 = (const float*)d_in[3];
  const float* b1 = (const float*)d_in[4];
  const float* W2 = (const float*)d_in[5];
  const float* b2 = (const float*)d_in[6];
  float* out = (float*)d_out;

  const int E  = in_sizes[2];
  const int Nn = in_sizes[0] / (kB * kFin * kT);

  // Workspace carve-out (~165 MB total for N=20000, E=320000)
  char* w = (char*)d_ws;
  auto carve = [&](size_t bytes) {
    void* p = (void*)w;
    w += (bytes + 255) & ~(size_t)255;
    return p;
  };
  float*    deg  = (float*)carve((size_t)Nn * 4);
  float*    dis  = (float*)carve((size_t)Nn * 4);
  float*    nrm  = (float*)carve((size_t)E * 4);
  _Float16* xT   = (_Float16*)carve((size_t)Nn * 512 * 2);   // [n][16][32] f16
  float*    aggX = (float*)carve((size_t)Nn * 512 * 4);      // [n][16][32] f32
  _Float16* h1   = (_Float16*)carve((size_t)Nn * 1024 * 2);  // [n][16][64] f16
  _Float16* g2   = (_Float16*)carve((size_t)Nn * 512 * 2);   // [n][16][32] f16
  float*    agg2 = (float*)carve((size_t)Nn * 512 * 4);      // [n][16][32] f32

  hipMemsetAsync(aggX, 0, (size_t)Nn * 512 * 4, stream);
  hipMemsetAsync(agg2, 0, (size_t)Nn * 512 * 4, stream);

  fill_f32<<<(Nn + 255) / 256, 256, 0, stream>>>(deg, 1.0f, Nn);       // self-loop wt
  deg_kernel<<<(E + 255) / 256, 256, 0, stream>>>(ei, ew, deg, E);
  dis_kernel<<<(Nn + 255) / 256, 256, 0, stream>>>(deg, dis, Nn);
  norm_kernel<<<(E + 255) / 256, 256, 0, stream>>>(ei, ew, dis, nrm, E);

  const int scatterWaves = (E + Nn + kEPW - 1) / kEPW;
  const int scatterBlocks = (scatterWaves + 7) / 8;
  const int gemmBlocks = (Nn + 8 * kNPW - 1) / (8 * kNPW);

  transpose_kernel<<<(Nn + 15) / 16, 256, 0, stream>>>(x, xT, Nn);
  scatter_kernel<<<scatterBlocks, 256, 0, stream>>>(xT, aggX, ei, nrm, dis, E, Nn);
  gemm1_kernel<<<gemmBlocks, 256, 0, stream>>>(aggX, W1, b1, h1, Nn);
  gemm2_kernel<<<gemmBlocks, 256, 0, stream>>>(h1, W2, g2, Nn);
  scatter_kernel<<<scatterBlocks, 256, 0, stream>>>(g2, agg2, ei, nrm, dis, E, Nn);
  final_kernel<<<(Nn + 15) / 16, 256, 0, stream>>>(agg2, b2, out, Nn);
}
